// MXMNet_67800353734793
// MI455X (gfx1250) — matrix operand order repaired and checked by reading, not compiled, not run on hardware
//
#include <hip/hip_runtime.h>
#include <math.h>

// ---------------------------------------------------------------------------
// MXMNet forward for MI455X (gfx1250), wave32 + WMMA bf16.
// One fused WMMA GEMM engine, templated on A-source mode and fixed-K so the
// hot paths (K=384 gather-concat edge MLPs, K=128 dense layers) are
// branch-free. A/B tiles are staged in LDS in *fragment-major* layout so each
// wave fetches its WMMA operands with 2x ds_load_b128 instead of 16 scalar
// 16-bit reads. Epilogue fuses bias+SiLU+(gathered) elementwise mul+residual
// +atomic scatter-add (segment_sum) so big intermediates touch HBM once.
// ---------------------------------------------------------------------------

#define DIM 128

typedef __attribute__((ext_vector_type(16))) __bf16 v16bf;
typedef __attribute__((ext_vector_type(8)))  float  v8f;

enum { F_BIAS=1, F_SILU=2, F_RES=4, F_MUL=8, F_MULG=16, F_SCAT=32 };

struct GArgs {
  const float* A;                       // plain A [M,K]   (AMODE 0)
  const float* P0; const int* i0;       // AMODE 1: concat(P0[i0[r]],P1[i1[r]],P2[r])
  const float* P1; const int* i1;
  const float* P2;
  const float* W;                       // [K,128] row major
  const float* bias;                    // [128]
  const float* addend;                  // F_RES   : += addend[row,col]
  const float* mul;                     // F_MUL   : *= mul[row,col]
  const float* mulg; const int* mulg_idx; // F_MULG: *= mulg[mulg_idx[row],col]
  const int*   scat;                    // F_SCAT  : atomicAdd(out[scat[row],col])
  float* out;
  long M; int K; int flags;
};

static __device__ inline unsigned short bf16bits(float f) {
  unsigned u = __builtin_bit_cast(unsigned, f);
  return (unsigned short)((u + 0x7FFFu + ((u >> 16) & 1u)) >> 16);  // RNE
}
static __device__ inline unsigned pack2(float a, float b) {
  return (unsigned)bf16bits(a) | ((unsigned)bf16bits(b) << 16);
}
static __device__ inline __bf16 bf1(float f) {
  unsigned short s = bf16bits(f);
  return __builtin_bit_cast(__bf16, s);
}

// LDS: fragment-major [slot][lane][16 bf16], per-lane slot padded to 24 elems
// (48B) to spread banks. 8 slots (waves / n-tiles) x 32 lanes.
#define FSTR 24

union Frag { v16bf v; uint4 q[2]; };

template<int AMODE, int KFIX>
__global__ __launch_bounds__(256) void k_gemm_t(GArgs g)
{
  __shared__ __align__(16) __bf16 sA[8 * 32 * FSTR];
  __shared__ __align__(16) __bf16 sB[8 * 32 * FSTR];
  const int tid  = threadIdx.x;
  const int wave = tid >> 5;            // 0..7 -> M strip of 16 rows
  const int lane = tid & 31;
  const long rowBase = (long)blockIdx.x * 128;
  const int K = KFIX ? KFIX : g.K;

  v8f acc[8];
#pragma unroll
  for (int t = 0; t < 8; ++t)
#pragma unroll
    for (int e = 0; e < 8; ++e) acc[t][e] = 0.f;

  // A staging coords: thread covers rows r = 32p + tr, cols tc..tc+3
  const int tr = tid >> 3;              // 0..31
  const int tc = (tid & 7) << 2;        // 0,4,...,28
  // destination for an A value at (row r, k-in-tile kk=tc..tc+3):
  //   16-bit A 16x32 ISA layout -> lane = (r&15) + 16*((kk>>3)&1),
  //   elem = (kk&7) + 8*(kk>>4)  (contiguous across the 4 cols)
  const int a_hi  = (tc >> 3) & 1;
  const int a_e   = (tc & 7) + ((tc >> 4) << 3);

  // B staging coords: thread covers rows r = 8p + br of W-tile, cols bc..bc+3
  const int br = tid >> 5;              // 0..7
  const int bc = lane << 2;             // 0..124
  const int b_nt = bc >> 4;
  const int b_ln = bc & 15;

  // preload gather indices once (AMODE 1)
  int gi0[4], gi1[4];
  if (AMODE == 1) {
#pragma unroll
    for (int p = 0; p < 4; ++p) {
      const long grow = rowBase + (p << 5) + tr;
      gi0[p] = g.i0[grow];
      gi1[p] = g.i1[grow];
    }
  }

  const int kTiles = (K + 31) >> 5;
  for (int kt = 0; kt < kTiles; ++kt) {
    const int k0 = kt << 5;

    // ---- stage A tile (f32 -> bf16, fragment-major) ----
#pragma unroll
    for (int p = 0; p < 4; ++p) {
      const int r = (p << 5) + tr;
      const long grow = rowBase + r;
      float4 v;
      if (AMODE == 1) {
        const int chunk = k0 >> 7;             // uniform across block
        const int kin   = (k0 & 127) + tc;
        const float* base = (chunk == 0) ? g.P0 + (long)gi0[p] * DIM
                          : (chunk == 1) ? g.P1 + (long)gi1[p] * DIM
                                         : g.P2 + grow * (long)DIM;
        v = *(const float4*)(base + kin);
      } else if (KFIX) {
        v = *(const float4*)(g.A + grow * (long)KFIX + k0 + tc);
        if (kt + 1 < kTiles)
          __builtin_prefetch(g.A + grow * (long)KFIX + k0 + 32 + tc, 0, 1);
      } else {
        const float* src = g.A + grow * (long)K;
        const int k = k0 + tc;
        v.x = (k + 0 < K) ? src[k + 0] : 0.f;
        v.y = (k + 1 < K) ? src[k + 1] : 0.f;
        v.z = (k + 2 < K) ? src[k + 2] : 0.f;
        v.w = (k + 3 < K) ? src[k + 3] : 0.f;
      }
      const int slot = ((r >> 4) << 5) + (r & 15) + (a_hi << 4);
      uint2 q; q.x = pack2(v.x, v.y); q.y = pack2(v.z, v.w);
      *(uint2*)&sA[slot * FSTR + a_e] = q;
    }
    // ---- stage B tile: W rows k0..k0+31 (fragment-major) ----
#pragma unroll
    for (int p = 0; p < 4; ++p) {
      const int r = (p << 3) + br;               // 0..31
      const int k = k0 + r;
      float4 v;
      if (KFIX || k < K) {
        v = *(const float4*)(g.W + (long)k * DIM + bc);
        if (kt + 1 < kTiles)
          __builtin_prefetch(g.W + (long)(k + 32) * DIM + bc, 0, 1);
      } else { v.x = v.y = v.z = v.w = 0.f; }
      // B 32x16 ISA layout: lane = col&15 (+16 if (k>>3)&1), elem = (k&7) + 8*(k>>4)
      const int slot = (b_nt << 5) + b_ln + (((r >> 3) & 1) << 4);
      __bf16* d = &sB[slot * FSTR + (r & 7) + ((r >> 4) << 3)];
      // 4 cols -> 4 consecutive lane-slots, same elem
      d[0 * FSTR] = bf1(v.x); d[1 * FSTR] = bf1(v.y);
      d[2 * FSTR] = bf1(v.z); d[3 * FSTR] = bf1(v.w);
    }
    __syncthreads();

    // ---- fragments: 2x b128 LDS loads each ----
    Frag af;
    {
      const uint4* pa = (const uint4*)&sA[((wave << 5) + lane) * FSTR];
      af.q[0] = pa[0]; af.q[1] = pa[1];
    }
#pragma unroll
    for (int nt = 0; nt < 8; ++nt) {
      Frag bfr;
      const uint4* pb = (const uint4*)&sB[((nt << 5) + lane) * FSTR];
      bfr.q[0] = pb[0]; bfr.q[1] = pb[1];
      acc[nt] = __builtin_amdgcn_wmma_f32_16x16x32_bf16(
          false, af.v, false, bfr.v, (short)0, acc[nt], false, false);
    }
    __syncthreads();
  }

  // ---- epilogue (C layout: VGPR r -> row r / r+8; lane&15 -> col) ----
  const int rlo = (wave << 4) + ((lane < 16) ? 0 : 8);
  const int cl  = lane & 15;
#pragma unroll
  for (int nt = 0; nt < 8; ++nt) {
    const int col = (nt << 4) + cl;
    const float bia = (g.flags & F_BIAS) ? g.bias[col] : 0.f;
#pragma unroll
    for (int r = 0; r < 8; ++r) {
      const long row = rowBase + rlo + r;
      float v = acc[nt][r] + bia;
      if (g.flags & F_SILU) v = v * (1.f / (1.f + __expf(-v)));
      if (g.flags & F_MUL)  v *= g.mul[row * DIM + col];
      if (g.flags & F_MULG) v *= g.mulg[(long)g.mulg_idx[row] * DIM + col];
      if (g.flags & F_RES)  v += g.addend[row * DIM + col];
      if (g.flags & F_SCAT) atomicAdd(&g.out[(long)g.scat[row] * DIM + col], v);
      else                  g.out[row * DIM + col] = v;
    }
  }
}

// ---------------------------------------------------------------------------
// Small utility / feature kernels
// ---------------------------------------------------------------------------
__global__ void k_zero(float* p, long n) {
  long i = (long)blockIdx.x * 256 + threadIdx.x;
  const long st = (long)gridDim.x * 256;
  for (; i < n; i += st) p[i] = 0.f;
}
__global__ void k_copy(float* d, const float* s, long n) {
  long i = (long)blockIdx.x * 256 + threadIdx.x;
  const long st = (long)gridDim.x * 256;
  for (; i < n; i += st) d[i] = s[i];
}
__global__ void k_embed(float* h, const float* emb, const int* x, long total) {
  long i = (long)blockIdx.x * 256 + threadIdx.x;
  if (i >= total) return;
  long n = i >> 7; int c = (int)(i & 127);
  h[i] = emb[(long)x[n] * DIM + c];
}
__global__ void k_dist(float* d, const float* pos, const int* a, const int* b, int E) {
  int e = blockIdx.x * 256 + threadIdx.x;
  if (e >= E) return;
  int ia = a[e], ib = b[e];
  float dx = pos[3*ia+0] - pos[3*ib+0];
  float dy = pos[3*ia+1] - pos[3*ib+1];
  float dz = pos[3*ia+2] - pos[3*ib+2];
  d[e] = sqrtf(dx*dx + dy*dy + dz*dz);
}
__global__ void k_angle(float* ang, const float* pos, const int* ia, const int* ib,
                        const int* ic, int T) {
  int t = blockIdx.x * 256 + threadIdx.x;
  if (t >= T) return;
  int a = ia[t], b = ib[t], c = ic[t];
  float v1x = pos[3*b+0]-pos[3*a+0], v1y = pos[3*b+1]-pos[3*a+1], v1z = pos[3*b+2]-pos[3*a+2];
  float v2x = pos[3*c+0]-pos[3*b+0], v2y = pos[3*c+1]-pos[3*b+1], v2z = pos[3*c+2]-pos[3*b+2];
  float cx = v1y*v2z - v1z*v2y, cy = v1z*v2x - v1x*v2z, cz = v1x*v2y - v1y*v2x;
  float dot = v1x*v2x + v1y*v2y + v1z*v2z;
  ang[t] = atan2f(sqrtf(cx*cx + cy*cy + cz*cz), dot);
}
__device__ inline float d_env(float x) {  // ENV_P = 5
  if (x >= 1.f) return 0.f;
  float xp = x*x*x*x*x;
  return 1.f/x - 21.f*xp + 35.f*xp*x - 15.f*xp*x*x;
}
__global__ void k_rbf_raw(float* out, const float* dist, const float* freq,
                          float inv_cutoff, int E) {
  int e = blockIdx.x * 256 + threadIdx.x;
  if (e >= E) return;
  float d = dist[e] * inv_cutoff;
  float env = d_env(d);
  for (int f = 0; f < 16; ++f)
    out[(long)e * 16 + f] = env * __sinf(freq[f] * d);
}
__device__ double dev_sph_jn(int n, double x) {
  double j0 = sin(x) / x;
  if (n == 0) return j0;
  double j1 = sin(x)/(x*x) - cos(x)/x;
  double jm2 = j0, jm1 = j1;
  for (int m = 2; m <= n; ++m) { double t = (2.0*m - 1.0)/x*jm1 - jm2; jm2 = jm1; jm1 = t; }
  return jm1;
}
// C layout: [0..41]=Z[7][6], [42..83]=NORM[7][6], [84..90]=SPH_PREF[7]
__global__ void k_init_consts(float* C) {
  if (threadIdx.x != 0 || blockIdx.x != 0) return;
  const double PI = 3.14159265358979323846;
  double zeros[7][6];
  for (int k = 0; k < 6; ++k) zeros[0][k] = (k + 1) * PI;
  double points[12]; int np = 12;
  for (int i = 0; i < 12; ++i) points[i] = (i + 1) * PI;
  for (int order = 1; order < 7; ++order) {
    double roots[12]; int nr = np - 1;
    for (int jj = 0; jj < nr; ++jj) {
      double a = points[jj], b = points[jj + 1];
      double fa = dev_sph_jn(order, a);
      for (int it = 0; it < 100; ++it) {
        double m = 0.5 * (a + b);
        double fm = dev_sph_jn(order, m);
        if ((fm >= 0.0) == (fa >= 0.0)) { a = m; fa = fm; } else { b = m; }
      }
      roots[jj] = 0.5 * (a + b);
    }
    for (int i = 0; i < nr; ++i) points[i] = roots[i];
    np = nr;
    for (int k = 0; k < 6; ++k) zeros[order][k] = points[k];
  }
  for (int n = 0; n < 7; ++n)
    for (int k = 0; k < 6; ++k) {
      C[n*6 + k] = (float)zeros[n][k];
      double jn1 = dev_sph_jn(n + 1, zeros[n][k]);
      C[42 + n*6 + k] = (float)(1.0 / sqrt(0.5 * jn1 * jn1));
    }
  for (int l = 0; l < 7; ++l) C[84 + l] = (float)sqrt((2.0*l + 1.0) / (4.0 * PI));
}
// per local edge: radial SBF part [E,42]: env(x) * j_n(Z[n,k]*x) * NORM[n,k]
__global__ void k_edge_sbf(float* out, const float* dist, const float* C, int E) {
  int e = blockIdx.x * 256 + threadIdx.x;
  if (e >= E) return;
  float x = dist[e] * (1.f / 5.f);
  float env = d_env(x);
  for (int n = 0; n < 7; ++n)
    for (int k = 0; k < 6; ++k) {
      float arg = C[n*6 + k] * x;
      float s = __sinf(arg), c = __cosf(arg);
      float j0 = s / arg, val;
      if (n == 0) val = j0;
      else {
        float j1 = s/(arg*arg) - c/arg;
        float jm2 = j0, jm1 = j1;
        for (int m = 2; m <= n; ++m) { float t = (2*m - 1)/arg*jm1 - jm2; jm2 = jm1; jm1 = t; }
        val = jm1;
      }
      out[(long)e*42 + n*6 + k] = env * val * C[42 + n*6 + k];
    }
}
// per triplet: sbf[t,42] = edge_sbf[idx_edge[t]] * (P_n(cos angle) * PREF[n])
__global__ void k_sbf_tri(float* out, const float* ang, const int* idxe,
                          const float* esbf, const float* C, int T) {
  int t = blockIdx.x * 256 + threadIdx.x;
  if (t >= T) return;
  float z = __cosf(ang[t]);
  float P[7]; P[0] = 1.f; P[1] = z;
  for (int l = 2; l < 7; ++l) P[l] = ((2*l - 1)*z*P[l-1] - (l - 1)*P[l-2]) / (float)l;
  long e = idxe[t];
  for (int n = 0; n < 7; ++n) {
    float cb = P[n] * C[84 + n];
    for (int k = 0; k < 6; ++k)
      out[(long)t*42 + n*6 + k] = esbf[e*42 + n*6 + k] * cb;
  }
}
// node_sum contribution: out[batch[n]] += dot(y[n,:], Wy) + by
__global__ void k_ydot(const float* y, const float* Wy, const float* by,
                       const int* batch, float* out, int N) {
  int n = blockIdx.x * 256 + threadIdx.x;
  if (n >= N) return;
  float s = 0.f;
  for (int c = 0; c < DIM; ++c) s += y[(long)n * DIM + c] * Wy[c];
  atomicAdd(&out[batch[n]], s + by[0]);
}

// ---------------------------------------------------------------------------
// Host orchestration
// ---------------------------------------------------------------------------
static inline void gemm(hipStream_t st, long M, int K, int amode, int flags,
                        const float* A, const float* P0, const int* i0,
                        const float* P1, const int* i1, const float* P2,
                        const float* W, const float* bias, float* out,
                        const float* addend = nullptr, const float* mul = nullptr,
                        const float* mulg = nullptr, const int* mulg_idx = nullptr,
                        const int* scat = nullptr) {
  GArgs a; a.A=A; a.P0=P0; a.i0=i0; a.P1=P1; a.i1=i1; a.P2=P2; a.W=W; a.bias=bias;
  a.addend=addend; a.mul=mul; a.mulg=mulg; a.mulg_idx=mulg_idx; a.scat=scat;
  a.out=out; a.M=M; a.K=K; a.flags=flags;
  dim3 grid((unsigned)((M + 127) / 128)), blkd(256);
  if (amode == 1)      k_gemm_t<1, 384><<<grid, blkd, 0, st>>>(a);
  else if (K == 128)   k_gemm_t<0, 128><<<grid, blkd, 0, st>>>(a);
  else                 k_gemm_t<0, 0  ><<<grid, blkd, 0, st>>>(a);
}
static inline void zerob(hipStream_t st, float* p, long n) {
  long b = (n + 255) / 256; if (b > 8192) b = 8192;
  k_zero<<<dim3((unsigned)b), dim3(256), 0, st>>>(p, n);
}
static inline void copyb(hipStream_t st, float* d, const float* s, long n) {
  long b = (n + 255) / 256; if (b > 8192) b = 8192;
  k_copy<<<dim3((unsigned)b), dim3(256), 0, st>>>(d, s, n);
}

// params pytree leaf indices (JAX dict flattening = alphabetical keys)
#define P_EMB 0
#define P_FREQ_G 1
#define P_FREQ_L 2
#define GBASE(L) (3 + 19 * (L))     // h_mlp(2) linear(1) mlp(2) res1(4) res2(4) res3(4) x_edge(2)
#define LBASE(L) (60 + 43 * (L))
#define P_RBF_G_W 189
#define P_RBF_G_B 190
#define P_RBF_L_W 191
#define P_RBF_L_B 192
#define P_SBF1_W 193
#define P_SBF1_B 194
#define P_SBF2_W 195
#define P_SBF2_B 196
#define I_POS 197
#define I_X 198
#define I_EIL 199
#define I_EIG 200
#define I_BATCH 201
#define I_I1 202
#define I_J 203
#define I_K 204
#define I_KJ 205
#define I_JI1 206
#define I_I2 207
#define I_J1 208
#define I_J2 209
#define I_JJ 210
#define I_JI2 211

extern "C" void kernel_launch(void* const* d_in, const int* in_sizes, int n_in,
                              void* d_out, int out_size, void* d_ws, size_t ws_size,
                              hipStream_t stream) {
  (void)n_in; (void)ws_size;
  const float* LF[213]; const int* LI[213];
  for (int i = 0; i < 213; ++i) { LF[i] = (const float*)d_in[i]; LI[i] = (const int*)d_in[i]; }

  const long N  = in_sizes[I_POS] / 3;
  const long EL = in_sizes[I_EIL] / 2;
  const long EG = in_sizes[I_EIG] / 2;
  const long T1 = in_sizes[I_KJ];
  const long T2 = in_sizes[I_JJ];

  const float* pos = LF[I_POS];
  const int* xatt  = LI[I_X];
  const int* jl = LI[I_EIL];           // edge_index_l[0]
  const int* il = LI[I_EIL] + EL;      // edge_index_l[1]  (local aggregates at i=row1)
  const int* ig = LI[I_EIG];           // edge_index_g[0]  (global aggregates at i=row0)
  const int* jg = LI[I_EIG] + EG;      // edge_index_g[1]
  const int* batch = LI[I_BATCH];
  float* out = (float*)d_out;

  // ---- bump-allocate workspace ----
  char* wp = (char*)d_ws;
  auto alloc = [&](long nflt) -> float* {
    float* p = (float*)wp;
    wp += ((nflt * 4 + 255) / 256) * 256;
    return p;
  };
  float* C       = alloc(128);
  float* distL   = alloc(EL);
  float* distG   = alloc(EG);
  float* ang1    = alloc(T1);
  float* ang2    = alloc(T2);
  float* rbfLraw = alloc(EL * 16);
  float* rbfGraw = alloc(EG * 16);
  float* esbf    = alloc(EL * 42);
  float* sbf1raw = alloc(T1 * 42);
  float* sbf2raw = alloc(T2 * 42);
  float* rbfL    = alloc(EL * DIM);
  float* rbfG    = alloc(EG * DIM);
  float* sbf1p   = alloc(T1 * DIM);
  float* sbf2p   = alloc(T2 * DIM);
  float* linE    = alloc(EG * DIM);
  float* nb[5];  for (int i = 0; i < 5; ++i) nb[i] = alloc(N * DIM);
  float* eA = alloc(EL * DIM); float* eB = alloc(EL * DIM);
  float* eC = alloc(EL * DIM); float* eD = alloc(EL * DIM);
  float* eE = alloc(EL * DIM);
  float* tb = alloc(T1 * DIM);

  auto blk = [](long n) { return dim3((unsigned)((n + 255) / 256)); };

  // ---- preamble: constants, geometry, basis features ----
  k_init_consts<<<dim3(1), dim3(1), 0, stream>>>(C);
  zerob(stream, out, out_size);
  k_embed<<<blk(N * DIM), dim3(256), 0, stream>>>(nb[0], LF[P_EMB], xatt, N * DIM);
  k_dist<<<blk(EL), dim3(256), 0, stream>>>(distL, pos, il, jl, (int)EL);
  k_dist<<<blk(EG), dim3(256), 0, stream>>>(distG, pos, ig, jg, (int)EG);
  k_angle<<<blk(T1), dim3(256), 0, stream>>>(ang1, pos, LI[I_I1], LI[I_J], LI[I_K], (int)T1);
  k_angle<<<blk(T2), dim3(256), 0, stream>>>(ang2, pos, LI[I_I2], LI[I_J1], LI[I_J2], (int)T2);
  k_rbf_raw<<<blk(EG), dim3(256), 0, stream>>>(rbfGraw, distG, LF[P_FREQ_G], 0.2f, (int)EG);
  k_rbf_raw<<<blk(EL), dim3(256), 0, stream>>>(rbfLraw, distL, LF[P_FREQ_L], 0.2f, (int)EL);
  gemm(stream, EG, 16, 0, F_BIAS | F_SILU, rbfGraw, 0, 0, 0, 0, 0, LF[P_RBF_G_W], LF[P_RBF_G_B], rbfG);
  gemm(stream, EL, 16, 0, F_BIAS | F_SILU, rbfLraw, 0, 0, 0, 0, 0, LF[P_RBF_L_W], LF[P_RBF_L_B], rbfL);
  k_edge_sbf<<<blk(EL), dim3(256), 0, stream>>>(esbf, distL, C, (int)EL);
  k_sbf_tri<<<blk(T1), dim3(256), 0, stream>>>(sbf1raw, ang1, LI[I_KJ], esbf, C, (int)T1);
  k_sbf_tri<<<blk(T2), dim3(256), 0, stream>>>(sbf2raw, ang2, LI[I_JJ], esbf, C, (int)T2);
  gemm(stream, T1, 42, 0, F_BIAS | F_SILU, sbf1raw, 0, 0, 0, 0, 0, LF[P_SBF1_W], LF[P_SBF1_B], sbf1p);
  gemm(stream, T2, 42, 0, F_BIAS | F_SILU, sbf2raw, 0, 0, 0, 0, 0, LF[P_SBF2_W], LF[P_SBF2_B], sbf2p);

  int hi = 0;
  for (int L = 0; L < 3; ++L) {
    // =============== global message passing ===============
    {
      int o[4], c = 0;
      for (int p = 0; p < 5; ++p) if (p != hi) o[c++] = p;
      float* h  = nb[hi];
      float* s0 = nb[o[0]]; float* s1 = nb[o[1]]; float* s2 = nb[o[2]];
      float* hO = nb[o[3]];
      const int G = GBASE(L);
      // lin_e = edge_attr @ linear
      gemm(stream, EG, DIM, 0, 0, rbfG, 0, 0, 0, 0, 0, LF[G+2], 0, linE);
      // h1 = silu(h @ h_mlp)
      gemm(stream, N, DIM, 0, F_BIAS | F_SILU, h, 0, 0, 0, 0, 0, LF[G+0], LF[G+1], s0);
      // prop: s1 = h1 + segsum_i( lin_e * silu([h1[i]|h1[j]|rbfG] @ x_edge) )
      copyb(stream, s1, s0, N * DIM);
      gemm(stream, EG, 384, 1, F_BIAS | F_SILU | F_MUL | F_SCAT,
           0, s0, ig, s0, jg, rbfG, LF[G+17], LF[G+18], s1, 0, linE, 0, 0, ig);
      // res1
      gemm(stream, N, DIM, 0, F_BIAS | F_SILU, s1, 0, 0, 0, 0, 0, LF[G+5], LF[G+6], s2);
      gemm(stream, N, DIM, 0, F_BIAS | F_SILU | F_RES, s2, 0, 0, 0, 0, 0, LF[G+7], LF[G+8], s0, s1);
      // mlp + res_h
      gemm(stream, N, DIM, 0, F_BIAS | F_SILU | F_RES, s0, 0, 0, 0, 0, 0, LF[G+3], LF[G+4], s1, h);
      // res2, res3
      gemm(stream, N, DIM, 0, F_BIAS | F_SILU, s1, 0, 0, 0, 0, 0, LF[G+9],  LF[G+10], s2);
      gemm(stream, N, DIM, 0, F_BIAS | F_SILU | F_RES, s2, 0, 0, 0, 0, 0, LF[G+11], LF[G+12], s0, s1);
      gemm(stream, N, DIM, 0, F_BIAS | F_SILU, s0, 0, 0, 0, 0, 0, LF[G+13], LF[G+14], s2);
      gemm(stream, N, DIM, 0, F_BIAS | F_SILU | F_RES, s2, 0, 0, 0, 0, 0, LF[G+15], LF[G+16], s1, s0);
      // prop again -> hO
      copyb(stream, hO, s1, N * DIM);
      gemm(stream, EG, 384, 1, F_BIAS | F_SILU | F_MUL | F_SCAT,
           0, s1, ig, s1, jg, rbfG, LF[G+17], LF[G+18], hO, 0, linE, 0, 0, ig);
      hi = o[3];
    }
    // =============== local message passing ===============
    {
      int o[4], c = 0;
      for (int p = 0; p < 5; ++p) if (p != hi) o[c++] = p;
      float* h  = nb[hi];
      float* s0 = nb[o[0]]; float* s1 = nb[o[1]]; float* s2 = nb[o[2]];
      float* hO = nb[o[3]];
      const int Lb = LBASE(L);
      // h1
      gemm(stream, N, DIM, 0, F_BIAS | F_SILU, h, 0, 0, 0, 0, 0, LF[Lb+0], LF[Lb+1], s0);
      // ekj = silu([h1[i]|h1[j]|rbfL] @ mlp_kj) * (rbfL @ lin_rbf1)
      gemm(stream, EL, DIM, 0, 0, rbfL, 0, 0, 0, 0, 0, LF[Lb+2], 0, eA);
      gemm(stream, EL, 384, 1, F_BIAS | F_SILU | F_MUL,
           0, s0, il, s0, jl, rbfL, LF[Lb+13], LF[Lb+14], eB, 0, eA);
      // m_kj = segsum_{ji1}( ekj[idx_kj] * sbf1_mlp(sbf1) )
      gemm(stream, T1, DIM, 0, F_BIAS | F_SILU, sbf1p, 0, 0, 0, 0, 0, LF[Lb+15], LF[Lb+16], tb);
      zerob(stream, eC, EL * DIM);
      gemm(stream, T1, DIM, 0, F_BIAS | F_SILU | F_MULG | F_SCAT,
           tb, 0, 0, 0, 0, 0, LF[Lb+17], LF[Lb+18], eC, 0, 0, eB, LI[I_KJ], LI[I_JI1]);
      // m1 = silu([h1[i]|h1[j]|rbfL] @ mlp_ji_1) + m_kj
      gemm(stream, EL, 384, 1, F_BIAS | F_SILU | F_RES,
           0, s0, il, s0, jl, rbfL, LF[Lb+7], LF[Lb+8], eD, eC);
      // ejj = silu(m1 @ mlp_jj) * (rbfL @ lin_rbf2)
      gemm(stream, EL, DIM, 0, 0, rbfL, 0, 0, 0, 0, 0, LF[Lb+3], 0, eA);
      gemm(stream, EL, DIM, 0, F_BIAS | F_SILU | F_MUL, eD, 0, 0, 0, 0, 0, LF[Lb+11], LF[Lb+12], eB, 0, eA);
      // m_jj = segsum_{ji2}( ejj[idx_jj] * sbf2_mlp(sbf2) )
      gemm(stream, T2, DIM, 0, F_BIAS | F_SILU, sbf2p, 0, 0, 0, 0, 0, LF[Lb+19], LF[Lb+20], tb);
      zerob(stream, eC, EL * DIM);
      gemm(stream, T2, DIM, 0, F_BIAS | F_SILU | F_MULG | F_SCAT,
           tb, 0, 0, 0, 0, 0, LF[Lb+21], LF[Lb+22], eC, 0, 0, eB, LI[I_JJ], LI[I_JI2]);
      // m2 = silu(m1 @ mlp_ji_2) + m_jj
      gemm(stream, EL, DIM, 0, F_BIAS | F_SILU | F_RES, eD, 0, 0, 0, 0, 0, LF[Lb+9], LF[Lb+10], eE, eC);
      // h_agg = segsum_i( (rbfL @ lin_rbf_out) * m2 )
      zerob(stream, s1, N * DIM);
      gemm(stream, EL, DIM, 0, F_MUL | F_SCAT, rbfL, 0, 0, 0, 0, 0, LF[Lb+4], 0, s1, 0, eE, 0, 0, il);
      // res1
      gemm(stream, N, DIM, 0, F_BIAS | F_SILU, s1, 0, 0, 0, 0, 0, LF[Lb+23], LF[Lb+24], s2);
      gemm(stream, N, DIM, 0, F_BIAS | F_SILU | F_RES, s2, 0, 0, 0, 0, 0, LF[Lb+25], LF[Lb+26], s0, s1);
      // mlp + res_h
      gemm(stream, N, DIM, 0, F_BIAS | F_SILU | F_RES, s0, 0, 0, 0, 0, 0, LF[Lb+5], LF[Lb+6], s1, h);
      // res2, res3 -> hO
      gemm(stream, N, DIM, 0, F_BIAS | F_SILU, s1, 0, 0, 0, 0, 0, LF[Lb+27], LF[Lb+28], s2);
      gemm(stream, N, DIM, 0, F_BIAS | F_SILU | F_RES, s2, 0, 0, 0, 0, 0, LF[Lb+29], LF[Lb+30], s0, s1);
      gemm(stream, N, DIM, 0, F_BIAS | F_SILU, s0, 0, 0, 0, 0, 0, LF[Lb+31], LF[Lb+32], s2);
      gemm(stream, N, DIM, 0, F_BIAS | F_SILU | F_RES, s2, 0, 0, 0, 0, 0, LF[Lb+33], LF[Lb+34], hO, s0);
      // y head: 3-layer MLP then out[batch[n]] += y·Wy + by
      gemm(stream, N, DIM, 0, F_BIAS | F_SILU, hO, 0, 0, 0, 0, 0, LF[Lb+37], LF[Lb+38], s1);
      gemm(stream, N, DIM, 0, F_BIAS | F_SILU, s1, 0, 0, 0, 0, 0, LF[Lb+39], LF[Lb+40], s2);
      gemm(stream, N, DIM, 0, F_BIAS | F_SILU, s2, 0, 0, 0, 0, 0, LF[Lb+41], LF[Lb+42], s0);
      k_ydot<<<blk(N), dim3(256), 0, stream>>>(s0, LF[Lb+35], LF[Lb+36], batch, out, (int)N);
      hi = o[3];
    }
  }
}
